// SetConv_12850542149715
// MI455X (gfx1250) — matrix operand order repaired and checked by reading, not compiled, not run on hardware
//
#include <hip/hip_runtime.h>

// ---------------------------------------------------------------------------
// CDNA5 (gfx1250, wave32) implementation of the EdgeConv/SetConv reference.
// Strategy: 4 recompute passes (stats1, stats2, stats3, final+maxpool), each
// built on v_wmma_f32_16x16x32_bf16 tiles. Tiny workspace (norm stats only).
// ---------------------------------------------------------------------------

typedef __attribute__((ext_vector_type(16))) __bf16 v16bf;
typedef __attribute__((ext_vector_type(8)))  __bf16 v8bf;
typedef __attribute__((ext_vector_type(8)))  float  v8f;

static constexpr int   Bn   = 4;
static constexpr int   Np   = 8192;            // points per batch
static constexpr int   CIN  = 32;
static constexpr int   Kn   = 32;              // neighbors per point
static constexpr int   PTS  = Bn * Np;         // 32768 points total
static constexpr int   CNT  = Np * Kn;         // reduction count per (b,ch)
static constexpr float EPSV = 1e-5f;
static constexpr float NEG  = 0.1f;

__device__ __forceinline__ v8f wmma_bf16(v16bf a, v16bf b, v8f c) {
  return __builtin_amdgcn_wmma_f32_16x16x32_bf16(false, a, false, b,
                                                 (short)0, c, false, false);
}

// B fragment (32x16 bf16): lane<16: elems 0..7 = K 0..7, 8..15 = K 16..23 of
// column n=lane; lane>=16: +8 on K (column n=lane-16). Weights in LDS [n][k].
__device__ __forceinline__ v16bf make_bfrag(const __bf16* __restrict__ w,
                                            int lane, int nt, int chunk) {
  const int n  = nt * 16 + (lane & 15);
  const int hi = lane >> 4;
  const __bf16* row = w + n * 64 + chunk * 32 + hi * 8;
  v8bf lo = *(const v8bf*)(row);
  v8bf up = *(const v8bf*)(row + 16);
  v16bf r;
#pragma unroll
  for (int i = 0; i < 8; ++i) { r[i] = lo[i]; r[i + 8] = up[i]; }
  return r;
}

// A fragment (16x32 bf16) from an LDS tile z[m][k] (row stride 64):
// lane<16: elems 0..7 = K 0..7, 8..15 = K 16..23 ; lane>=16: +8 on K.
__device__ __forceinline__ v16bf make_afrag_lds(const __bf16* __restrict__ z,
                                                int lane, int chunk) {
  const int m  = lane & 15;
  const int hi = lane >> 4;
  const __bf16* row = z + m * 64 + chunk * 32 + hi * 8;
  v8bf lo = *(const v8bf*)(row);        // K = base + 0..7
  v8bf up = *(const v8bf*)(row + 16);   // K = base + 16..23
  v16bf r;
#pragma unroll
  for (int i = 0; i < 8; ++i) { r[i] = lo[i]; r[i + 8] = up[i]; }
  return r;
}

// Layer-1 A fragment, K-chunk 0 (the 32 gathered signal channels), straight
// from the global gather row (fp32 -> bf16).
__device__ __forceinline__ v16bf make_afrag_gather(const float* __restrict__ srow,
                                                   int hi) {
  const float4 f0 = *(const float4*)(srow + hi * 8);
  const float4 f1 = *(const float4*)(srow + hi * 8 + 4);
  const float4 f2 = *(const float4*)(srow + 16 + hi * 8);
  const float4 f3 = *(const float4*)(srow + 16 + hi * 8 + 4);
  v16bf r;
  r[0] = (__bf16)f0.x; r[1] = (__bf16)f0.y; r[2]  = (__bf16)f0.z; r[3]  = (__bf16)f0.w;
  r[4] = (__bf16)f1.x; r[5] = (__bf16)f1.y; r[6]  = (__bf16)f1.z; r[7]  = (__bf16)f1.w;
  r[8] = (__bf16)f2.x; r[9] = (__bf16)f2.y; r[10] = (__bf16)f2.z; r[11] = (__bf16)f2.w;
  r[12] = (__bf16)f3.x; r[13] = (__bf16)f3.y; r[14] = (__bf16)f3.z; r[15] = (__bf16)f3.w;
  return r;
}

__device__ __forceinline__ void accum_stats(const v8f* y, int lane,
                                            float* s_sum, float* s_sq) {
  const int ml = lane & 15;
#pragma unroll
  for (int nt = 0; nt < 4; ++nt) {
    const int n = nt * 16 + ml;
    float s = 0.f, q = 0.f;
#pragma unroll
    for (int r = 0; r < 8; ++r) { const float v = y[nt][r]; s += v; q += v * v; }
    atomicAdd(&s_sum[n], s);
    atomicAdd(&s_sq[n], q);
  }
}

// z = leakyrelu(y * A + C), stored bf16 into the wave's 16x64 LDS tile.
__device__ __forceinline__ void affine_store(const v8f* y, int lane, int b,
                                             const float* __restrict__ A,
                                             const float* __restrict__ C,
                                             __bf16* __restrict__ zt) {
  const int ml = lane & 15;
  const int mb = (lane >> 4) * 8;
#pragma unroll
  for (int nt = 0; nt < 4; ++nt) {
    const int n = nt * 16 + ml;
    const float aa = A[b * 64 + n];
    const float cc = C[b * 64 + n];
#pragma unroll
    for (int r = 0; r < 8; ++r) {
      float z = y[nt][r] * aa + cc;
      z = (z >= 0.0f) ? z : NEG * z;
      zt[(mb + r) * 64 + n] = (__bf16)z;
    }
  }
}

template <int DEPTH>
__global__ void __launch_bounds__(256)
pass_kernel(const float* __restrict__ signal,
            const int*   __restrict__ edges,
            const float* __restrict__ efeat,
            const float* __restrict__ W1,
            const float* __restrict__ W2,
            const float* __restrict__ W3,
            const float* __restrict__ A1, const float* __restrict__ C1,
            const float* __restrict__ A2, const float* __restrict__ C2,
            const float* __restrict__ A3, const float* __restrict__ C3,
            float* __restrict__ gsum, float* __restrict__ gsumsq,
            float* __restrict__ out) {
  constexpr int NL = (DEPTH < 3) ? DEPTH : 3;
  __shared__ __align__(16) __bf16 wlds[NL][64 * 64];
  __shared__ __align__(16) __bf16 zt[8][16 * 64];
  __shared__ float s_sum[64];
  __shared__ float s_sq[64];

  const int tid  = threadIdx.x;
  const int wave = tid >> 5;
  const int lane = tid & 31;

  // Stage weights into LDS as bf16 [out][in], K padded to 64.
#pragma unroll
  for (int l = 0; l < NL; ++l) {
    const float* W = (l == 0) ? W1 : ((l == 1) ? W2 : W3);
    const int kd = (l == 0) ? (CIN + 3) : 64;
    for (int i = tid; i < 64 * 64; i += 256) {
      const int o = i >> 6, kk = i & 63;
      wlds[l][i] = (__bf16)((kk < kd) ? W[o * kd + kk] : 0.0f);
    }
  }
  if (DEPTH < 4 && tid < 64) { s_sum[tid] = 0.0f; s_sq[tid] = 0.0f; }
  __syncthreads();

  const int p  = blockIdx.x * 8 + wave;   // global point id (all waves same batch)
  const int b  = p / Np;
  const int hi = lane >> 4;
  const int ml = lane & 15;
  __bf16* myzt = zt[wave];

  float mx[4];
#pragma unroll
  for (int nt = 0; nt < 4; ++nt) mx[nt] = -3.0e38f;

#pragma unroll
  for (int t = 0; t < 2; ++t) {
    const int e = p * Kn + t * 16 + ml;   // this lane's edge (row m of the tile)

    // ---------------- layer 1: y = W1 * [gather(signal) ; edge_feats] -------
    const int idx = edges[e];
    const float* srow = signal + (size_t)idx * CIN;
    const v16bf a0 = make_afrag_gather(srow, hi);
    v16bf a1;
#pragma unroll
    for (int i = 0; i < 16; ++i) a1[i] = (__bf16)0.0f;
    if (hi == 0) {                        // K=32..34 are the 3 edge features
      a1[0] = (__bf16)efeat[e * 3 + 0];
      a1[1] = (__bf16)efeat[e * 3 + 1];
      a1[2] = (__bf16)efeat[e * 3 + 2];
    }
    v8f y[4];
#pragma unroll
    for (int nt = 0; nt < 4; ++nt) {
      v8f acc;
#pragma unroll
      for (int r = 0; r < 8; ++r) acc[r] = 0.0f;
      acc = wmma_bf16(a0, make_bfrag(wlds[0], lane, nt, 0), acc);
      acc = wmma_bf16(a1, make_bfrag(wlds[0], lane, nt, 1), acc);
      y[nt] = acc;
    }
    if constexpr (DEPTH == 1) { accum_stats(y, lane, s_sum, s_sq); continue; }

    // ---------------- layer 2 ----------------------------------------------
    if constexpr (DEPTH >= 2) {
      affine_store(y, lane, b, A1, C1, myzt);
      asm volatile("s_wait_dscnt 0" ::: "memory");   // order LDS store->load (same wave)
      const v16bf z0 = make_afrag_lds(myzt, lane, 0);
      const v16bf z1 = make_afrag_lds(myzt, lane, 1);
#pragma unroll
      for (int nt = 0; nt < 4; ++nt) {
        v8f acc;
#pragma unroll
        for (int r = 0; r < 8; ++r) acc[r] = 0.0f;
        acc = wmma_bf16(z0, make_bfrag(wlds[1], lane, nt, 0), acc);
        acc = wmma_bf16(z1, make_bfrag(wlds[1], lane, nt, 1), acc);
        y[nt] = acc;
      }
    }
    if constexpr (DEPTH == 2) { accum_stats(y, lane, s_sum, s_sq); continue; }

    // ---------------- layer 3 ----------------------------------------------
    if constexpr (DEPTH >= 3) {
      affine_store(y, lane, b, A2, C2, myzt);
      asm volatile("s_wait_dscnt 0" ::: "memory");
      const v16bf z0 = make_afrag_lds(myzt, lane, 0);
      const v16bf z1 = make_afrag_lds(myzt, lane, 1);
#pragma unroll
      for (int nt = 0; nt < 4; ++nt) {
        v8f acc;
#pragma unroll
        for (int r = 0; r < 8; ++r) acc[r] = 0.0f;
        acc = wmma_bf16(z0, make_bfrag(wlds[2], lane, nt, 0), acc);
        acc = wmma_bf16(z1, make_bfrag(wlds[2], lane, nt, 1), acc);
        y[nt] = acc;
      }
    }
    if constexpr (DEPTH == 3) { accum_stats(y, lane, s_sum, s_sq); continue; }

    // ---------------- final: norm+act layer 3, running max over neighbors ---
    if constexpr (DEPTH == 4) {
#pragma unroll
      for (int nt = 0; nt < 4; ++nt) {
        const int n = nt * 16 + ml;
        const float aa = A3[b * 64 + n];
        const float cc = C3[b * 64 + n];
#pragma unroll
        for (int r = 0; r < 8; ++r) {
          float z = y[nt][r] * aa + cc;
          z = (z >= 0.0f) ? z : NEG * z;
          mx[nt] = fmaxf(mx[nt], z);
        }
      }
    }
  }

  if constexpr (DEPTH < 4) {
    __syncthreads();
    if (tid < 64) {
      atomicAdd(&gsum[b * 64 + tid], s_sum[tid]);
      atomicAdd(&gsumsq[b * 64 + tid], s_sq[tid]);
    }
  } else {
    // lanes l and l+16 hold the same channel (different neighbor halves).
#pragma unroll
    for (int nt = 0; nt < 4; ++nt) {
      const float other = __shfl_xor(mx[nt], 16);
      mx[nt] = fmaxf(mx[nt], other);
    }
    if (lane < 16) {
      float* orow = out + (size_t)p * 64;   // output is (B, N, 64) row-major
#pragma unroll
      for (int nt = 0; nt < 4; ++nt) orow[nt * 16 + lane] = mx[nt];
    }
  }
}

__global__ void init_ws(float* ws) {
  const int i = blockIdx.x * blockDim.x + threadIdx.x;
  if (i < 3 * 1024) ws[i] = 0.0f;
}

__global__ void finalize_stats(const float* __restrict__ sum,
                               const float* __restrict__ sumsq,
                               const float* __restrict__ g,
                               const float* __restrict__ bb,
                               float* __restrict__ A, float* __restrict__ C) {
  const int i  = threadIdx.x;       // 256 = B*64
  const int ch = i & 63;
  const float inv_cnt = 1.0f / (float)CNT;
  const float mu  = sum[i] * inv_cnt;
  const float var = sumsq[i] * inv_cnt - mu * mu;
  const float inv = rsqrtf(var + EPSV);
  const float aa  = g[ch] * inv;
  A[i] = aa;
  C[i] = bb[ch] - mu * aa;
}

extern "C" void kernel_launch(void* const* d_in, const int* in_sizes, int n_in,
                              void* d_out, int out_size, void* d_ws, size_t ws_size,
                              hipStream_t stream) {
  const float* signal = (const float*)d_in[0];
  const int*   edges  = (const int*)d_in[1];
  const float* efeat  = (const float*)d_in[2];
  // d_in[3] = k (scalar, == 32, hardcoded)
  const float* W1 = (const float*)d_in[4];
  const float* g1 = (const float*)d_in[5];
  const float* b1 = (const float*)d_in[6];
  const float* W2 = (const float*)d_in[7];
  const float* g2 = (const float*)d_in[8];
  const float* b2 = (const float*)d_in[9];
  const float* W3 = (const float*)d_in[10];
  const float* g3 = (const float*)d_in[11];
  const float* b3 = (const float*)d_in[12];
  float* out = (float*)d_out;
  float* ws  = (float*)d_ws;

  // Per-layer stats region: [sum(256), sumsq(256), A(256), C(256)]
  float* sum1 = ws + 0 * 1024; float* sq1 = sum1 + 256; float* A1 = sum1 + 512; float* C1 = sum1 + 768;
  float* sum2 = ws + 1 * 1024; float* sq2 = sum2 + 256; float* A2 = sum2 + 512; float* C2 = sum2 + 768;
  float* sum3 = ws + 2 * 1024; float* sq3 = sum3 + 256; float* A3 = sum3 + 512; float* C3 = sum3 + 768;

  init_ws<<<12, 256, 0, stream>>>(ws);

  const dim3 grid(PTS / 8), blk(256);

  pass_kernel<1><<<grid, blk, 0, stream>>>(signal, edges, efeat, W1, W2, W3,
                                           A1, C1, A2, C2, A3, C3, sum1, sq1, out);
  finalize_stats<<<1, 256, 0, stream>>>(sum1, sq1, g1, b1, A1, C1);

  pass_kernel<2><<<grid, blk, 0, stream>>>(signal, edges, efeat, W1, W2, W3,
                                           A1, C1, A2, C2, A3, C3, sum2, sq2, out);
  finalize_stats<<<1, 256, 0, stream>>>(sum2, sq2, g2, b2, A2, C2);

  pass_kernel<3><<<grid, blk, 0, stream>>>(signal, edges, efeat, W1, W2, W3,
                                           A1, C1, A2, C2, A3, C3, sum3, sq3, out);
  finalize_stats<<<1, 256, 0, stream>>>(sum3, sq3, g3, b3, A3, C3);

  pass_kernel<4><<<grid, blk, 0, stream>>>(signal, edges, efeat, W1, W2, W3,
                                           A1, C1, A2, C2, A3, C3, nullptr, nullptr, out);
}
